// xLSTMCell_65962107732595
// MI455X (gfx1250) — matrix operand order repaired and checked by reading, not compiled, not run on hardware
//
#include <hip/hip_runtime.h>

// xLSTM cell, fused. Fast path:
//   1) one-shot pack+convert into d_ws:  A2 = [x | h_prev]  (4096 x 2048 bf16)
//      Wc[k] = [W_k | U_k]              (6 x 1024 x 2048 bf16)
//   2) 7 accumulation passes per 128x128 output tile; each z-gate is a single
//      K=2048 bf16 WMMA GEMM; gate 4's wx/uh are the two K-halves of panel 4.
// Staging: GLOBAL_LOAD_ASYNC_TO_LDS_B128, 4-deep LDS pipeline with partial
// ASYNCcnt waits (s_wait_asynccnt 4) so DMA runs ~2 K-steps ahead of the WMMAs.
// Fallback (ws too small): fp32-input kernel converting inside the loop.

typedef __attribute__((ext_vector_type(16))) __bf16 v16bf;
typedef __attribute__((ext_vector_type(8)))  __bf16 v8bf;
typedef __attribute__((ext_vector_type(8)))  float  v8f;
typedef __attribute__((ext_vector_type(4)))  int    v4i;

#define B_DIM 4096
#define HD    1024              // D_IN == H == 1024
#define HD2   2048              // packed K (x|h), packed ld
#define HH    (1024 * 1024)
#define PH    (1024 * 2048)     // packed panel stride (rows=1024, ld=2048)
#define BM    128
#define BN    128
#define KT    32                // K step == WMMA K
#define NT    4                 // n-tiles per wave (16 x 64 strip)
#define TSZ   (BM * KT)         // one LDS tile buffer (4096 elems = 8 KB)
#define NBUF  4                 // pipeline depth
#define THREADS 512             // 16 wave32s: 8 m-strips x 2 n-halves

#if defined(__gfx1250__) && __has_builtin(__builtin_amdgcn_global_load_async_to_lds_b128)
#define USE_ASYNC_LDS 1
typedef __attribute__((address_space(1))) v4i as1_v4i;   // global
typedef __attribute__((address_space(3))) v4i as3_v4i;   // LDS
#else
#define USE_ASYNC_LDS 0
#endif

__device__ __forceinline__ __bf16 f2bf(float f) {
  unsigned u = __builtin_bit_cast(unsigned, f);
  u += 0x7FFFu + ((u >> 16) & 1u);               // round-to-nearest-even
  unsigned short h = (unsigned short)(u >> 16);
  return __builtin_bit_cast(__bf16, h);
}

__device__ __forceinline__ v8bf cvt8(float4 a, float4 b) {
#if __has_builtin(__builtin_amdgcn_cvt_pk_bf16_f32)
  auto p0 = __builtin_amdgcn_cvt_pk_bf16_f32(a.x, a.y);
  auto p1 = __builtin_amdgcn_cvt_pk_bf16_f32(a.z, a.w);
  auto p2 = __builtin_amdgcn_cvt_pk_bf16_f32(b.x, b.y);
  auto p3 = __builtin_amdgcn_cvt_pk_bf16_f32(b.z, b.w);
  v8bf r;
  r[0] = p0[0]; r[1] = p0[1]; r[2] = p1[0]; r[3] = p1[1];
  r[4] = p2[0]; r[5] = p2[1]; r[6] = p3[0]; r[7] = p3[1];
  return r;
#else
  v8bf r;
  r[0] = f2bf(a.x); r[1] = f2bf(a.y); r[2] = f2bf(a.z); r[3] = f2bf(a.w);
  r[4] = f2bf(b.x); r[5] = f2bf(b.y); r[6] = f2bf(b.z); r[7] = f2bf(b.w);
  return r;
#endif
}

__device__ __forceinline__ float sigm(float v) {
  return 1.0f / (1.0f + __expf(-v));
}

// ---- pack+convert: src fp32 [rows,1024] -> dst bf16 [rows,2048] at colOff ----
__global__ __launch_bounds__(256)
void cvt_pack(const float* __restrict__ src, __bf16* __restrict__ dst,
              int n8, int colOff) {
  int i = blockIdx.x * blockDim.x + threadIdx.x;
  if (i >= n8) return;
  size_t e = (size_t)i * 8;
  size_t row = e >> 10, col = e & 1023;
  const float4* p = (const float4*)(src + e);
  *(v8bf*)(dst + row * HD2 + colOff + col) = cvt8(p[0], p[1]);
}

// barrier that drains the async-to-LDS queue down to CNT outstanding ops.
// CNT=4 keeps the 2 just-issued tile-pairs in flight (in-order completion
// guarantees everything older has landed); CNT=0 is a full drain.
template <int CNT>
__device__ __forceinline__ void sync_tiles() {
#if USE_ASYNC_LDS
#if __has_builtin(__builtin_amdgcn_s_wait_asynccnt)
  __builtin_amdgcn_s_wait_asynccnt(CNT);
#else
  asm volatile("s_wait_asynccnt %0" :: "i"(CNT) : "memory");
#endif
#endif
  __syncthreads();
}

// ---- stage a 128x32 tile (row-major, leading dim ld) into LDS as bf16 ----
// fp32 source: convert on the fly (fallback path)
__device__ __forceinline__ void load_tile(__bf16* lds, const float* __restrict__ src,
                                          int rowBase, int k0, int ld, int tid) {
  const int row = tid >> 2;                       // 0..127
  const int seg = (tid & 3) << 3;                 // 0,8,16,24
  const float* p = src + (size_t)(rowBase + row) * ld + k0 + seg;
  const float4* p4 = (const float4*)p;
  float4 f0 = p4[0], f1 = p4[1];
  *(v8bf*)&lds[row * KT + seg] = cvt8(f0, f1);
}

// bf16 source: async DMA straight into LDS (no VGPR round trip) when available
__device__ __forceinline__ void load_tile(__bf16* lds, const __bf16* __restrict__ src,
                                          int rowBase, int k0, int ld, int tid) {
  const int row = tid >> 2;
  const int seg = (tid & 3) << 3;
  const __bf16* p = src + (size_t)(rowBase + row) * ld + k0 + seg;
#if USE_ASYNC_LDS
  __builtin_amdgcn_global_load_async_to_lds_b128(
      (as1_v4i*)p, (as3_v4i*)&lds[row * KT + seg], 0, 0);
#else
  *(v8bf*)&lds[row * KT + seg] = *(const v8bf*)p;
#endif
}

// A fragment (16x32 bf16): lanes 0-15 row M=lane hold K{0-7,16-23},
// lanes 16-31 row M=lane-16 hold K{8-15,24-31}   (CDNA5 ISA 7.12.2)
__device__ __forceinline__ v16bf frag_a(const __bf16* lds, int rbase, int lane) {
  const int lr = lane & 15;
  const int kb = (lane >> 4) << 3;                // 0 or 8
  const __bf16* r = lds + (rbase + lr) * KT;
  v8bf lo = *(const v8bf*)(r + kb);
  v8bf hi = *(const v8bf*)(r + kb + 16);
  return __builtin_shufflevector(lo, hi, 0,1,2,3,4,5,6,7,8,9,10,11,12,13,14,15);
}

// B fragment (32x16 bf16): lanes 0-15 col N=lane hold K{0-7,16-23},
// lanes 16-31 col N=lane-16 hold K{8-15,24-31}
__device__ __forceinline__ v16bf frag_b(const __bf16* lds, int cbase, int lane) {
  const int lc = lane & 15;
  const int kb = (lane >> 4) << 3;                // 0 or 8
  const __bf16* r = lds + (cbase + lc) * KT;
  v8bf lo = *(const v8bf*)(r + kb);
  v8bf hi = *(const v8bf*)(r + kb + 16);
  return __builtin_shufflevector(lo, hi, 0,1,2,3,4,5,6,7,8,9,10,11,12,13,14,15);
}

__device__ __forceinline__ void do_wmma(v8f acc[NT], const __bf16* lA, const __bf16* lB,
                                        int lane, int mw, int nw) {
  v16bf af = frag_a(lA, mw * 16, lane);
#pragma unroll
  for (int t = 0; t < NT; ++t) {
    v16bf bfr = frag_b(lB, nw * 64 + t * 16, lane);
    acc[t] = __builtin_amdgcn_wmma_f32_16x16x32_bf16(
        false, af, false, bfr, (short)0, acc[t], false, false);
  }
}

// acc += A[mBase:+128, kBeg:kEnd] @ B[nBase:+128, kBeg:kEnd]^T
// 4-deep LDS pipeline; DMA runs 3 tiles ahead of consumption; stage count
// ((kEnd-kBeg)/KT = 32 or 64) is divisible by NBUF=4 so buffer offsets are
// compile-time constants. Tail stages issue clamped dummy loads to keep the
// ASYNCcnt cadence exact; a full drain ends the pass before buffer reuse.
template <typename T>
__device__ __forceinline__ void gemm_src(v8f acc[NT], __bf16* lA, __bf16* lB,
                                         const T* __restrict__ A,
                                         const T* __restrict__ Bw, int ld,
                                         int kBeg, int kEnd,
                                         int mBase, int nBase,
                                         int tid, int lane, int mw, int nw) {
  // prologue: stage 3 tiles ahead
#pragma unroll
  for (int s = 0; s < NBUF - 1; ++s) {
    load_tile(lA + s * TSZ, A,  mBase, kBeg + s * KT, ld, tid);
    load_tile(lB + s * TSZ, Bw, nBase, kBeg + s * KT, ld, tid);
  }
  sync_tiles<4>();                                 // tile 0 resident
  for (int k0 = kBeg; k0 < kEnd; k0 += NBUF * KT) {
#pragma unroll
    for (int s = 0; s < NBUF; ++s) {
      int kl = k0 + (s + NBUF - 1) * KT;           // load 3 tiles ahead
      if (kl >= kEnd) kl = kEnd - KT;              // tail: dummy (cadence only)
      load_tile(lA + ((s + NBUF - 1) & (NBUF - 1)) * TSZ, A,  mBase, kl, ld, tid);
      load_tile(lB + ((s + NBUF - 1) & (NBUF - 1)) * TSZ, Bw, nBase, kl, ld, tid);
      do_wmma(acc, lA + s * TSZ, lB + s * TSZ, lane, mw, nw);
      sync_tiles<4>();                             // next tile resident, 2 pairs fly
    }
  }
  sync_tiles<0>();                                 // full drain before buffer reuse
}

// ---------------- shared epilogue helpers ----------------
struct TileCtx {
  int tid, lane, mw, nw, mBase, nBase, colLo, rowOff, colBase;
};

__device__ __forceinline__ TileCtx make_ctx() {
  TileCtx c;
  c.tid  = threadIdx.x;
  c.lane = c.tid & 31;
  int wid = c.tid >> 5;
  c.mw   = wid & 7;
  c.nw   = wid >> 3;
  c.mBase = blockIdx.x * BM;
  c.nBase = blockIdx.y * BN;
  c.colLo = c.lane & 15;
  c.rowOff = c.mw * 16 + ((c.lane >> 4) << 3);
  c.colBase = c.nBase + c.nw * 64;
  return c;
}

// ---------------- fast path: packed bf16 kernel ----------------
__global__ __launch_bounds__(THREADS, 1)
void xlstm_packed(const __bf16* __restrict__ A2, const float* __restrict__ c_prev,
                  const __bf16* __restrict__ Wc, const float* __restrict__ b_all,
                  float* __restrict__ out)
{
  __shared__ __align__(16) __bf16 lA[NBUF * TSZ];   // 32 KB
  __shared__ __align__(16) __bf16 lB[NBUF * TSZ];   // 32 KB
  const TileCtx c = make_ctx();
  const v8f vz = {0.f, 0.f, 0.f, 0.f, 0.f, 0.f, 0.f, 0.f};
  v8f acc[NT], keepA[NT], keepB[NT];

#define GEMM(panel, kb, ke) \
  gemm_src(acc, lA, lB, A2, Wc + (size_t)(panel) * PH, HD2, kb, ke, \
           c.mBase, c.nBase, c.tid, c.lane, c.mw, c.nw)
#define ZERO() _Pragma("unroll") for (int t = 0; t < NT; ++t) acc[t] = vz

  // pass 1: z3 -> g = tanh
  ZERO(); GEMM(3, 0, HD2);
#pragma unroll
  for (int t = 0; t < NT; ++t) {
    float bb = b_all[3 * HD + c.colBase + t * 16 + c.colLo];
#pragma unroll
    for (int r = 0; r < 8; ++r) keepA[t][r] = tanhf(acc[t][r] + bb);
  }
  // pass 2: z0 -> i ; keepA = i*g
  ZERO(); GEMM(0, 0, HD2);
#pragma unroll
  for (int t = 0; t < NT; ++t) {
    float bb = b_all[c.colBase + t * 16 + c.colLo];
#pragma unroll
    for (int r = 0; r < 8; ++r) keepA[t][r] *= sigm(acc[t][r] + bb);
  }
  // pass 3: wx4 (x-half of panel 4) + b4
  ZERO(); GEMM(4, 0, HD);
#pragma unroll
  for (int t = 0; t < NT; ++t) {
    float bb = b_all[4 * HD + c.colBase + t * 16 + c.colLo];
#pragma unroll
    for (int r = 0; r < 8; ++r) keepB[t][r] = acc[t][r] + bb;
  }
  // pass 4: uh4 (h-half of panel 4) ; keepB = m
  ZERO(); GEMM(4, HD, HD2);
#pragma unroll
  for (int t = 0; t < NT; ++t)
#pragma unroll
    for (int r = 0; r < 8; ++r) keepB[t][r] *= acc[t][r];
  // pass 5: z5 -> a ; keepA += a*m
  ZERO(); GEMM(5, 0, HD2);
#pragma unroll
  for (int t = 0; t < NT; ++t) {
    float bb = b_all[5 * HD + c.colBase + t * 16 + c.colLo];
#pragma unroll
    for (int r = 0; r < 8; ++r) keepA[t][r] += sigm(acc[t][r] + bb) * keepB[t][r];
  }
  // pass 6: z1 -> f ; c_new = f*c_prev + i*g + a*m
  ZERO(); GEMM(1, 0, HD2);
#pragma unroll
  for (int t = 0; t < NT; ++t) {
    float bb = b_all[HD + c.colBase + t * 16 + c.colLo];
#pragma unroll
    for (int r = 0; r < 8; ++r) {
      size_t idx = (size_t)(c.mBase + c.rowOff + r) * HD + (c.colBase + t * 16 + c.colLo);
      keepA[t][r] += sigm(acc[t][r] + bb) * c_prev[idx];
      out[(size_t)B_DIM * HD + idx] = keepA[t][r];          // c_new
    }
  }
  // pass 7: z2 -> o ; h_new = o*tanh(c_new)
  ZERO(); GEMM(2, 0, HD2);
#pragma unroll
  for (int t = 0; t < NT; ++t) {
    float bb = b_all[2 * HD + c.colBase + t * 16 + c.colLo];
#pragma unroll
    for (int r = 0; r < 8; ++r) {
      size_t idx = (size_t)(c.mBase + c.rowOff + r) * HD + (c.colBase + t * 16 + c.colLo);
      out[idx] = sigm(acc[t][r] + bb) * tanhf(keepA[t][r]); // h_new
    }
  }
#undef GEMM
#undef ZERO
}

// ---------------- fallback: fp32 inputs, convert in-loop ----------------
__global__ __launch_bounds__(THREADS, 1)
void xlstm_f32(const float* __restrict__ x, const float* __restrict__ h_prev,
               const float* __restrict__ c_prev, const float* __restrict__ W_all,
               const float* __restrict__ b_all, const float* __restrict__ U_all,
               float* __restrict__ out)
{
  __shared__ __align__(16) __bf16 lA[NBUF * TSZ];
  __shared__ __align__(16) __bf16 lB[NBUF * TSZ];
  const TileCtx c = make_ctx();
  const v8f vz = {0.f, 0.f, 0.f, 0.f, 0.f, 0.f, 0.f, 0.f};
  v8f acc[NT], keepA[NT], keepB[NT];

#define GEMM(Asrc, Bsrc) \
  gemm_src(acc, lA, lB, Asrc, Bsrc, HD, 0, HD, \
           c.mBase, c.nBase, c.tid, c.lane, c.mw, c.nw)
#define ZERO() _Pragma("unroll") for (int t = 0; t < NT; ++t) acc[t] = vz

  ZERO(); GEMM(x, W_all + 3 * HH); GEMM(h_prev, U_all + 3 * HH);
#pragma unroll
  for (int t = 0; t < NT; ++t) {
    float bb = b_all[3 * HD + c.colBase + t * 16 + c.colLo];
#pragma unroll
    for (int r = 0; r < 8; ++r) keepA[t][r] = tanhf(acc[t][r] + bb);
  }
  ZERO(); GEMM(x, W_all); GEMM(h_prev, U_all);
#pragma unroll
  for (int t = 0; t < NT; ++t) {
    float bb = b_all[c.colBase + t * 16 + c.colLo];
#pragma unroll
    for (int r = 0; r < 8; ++r) keepA[t][r] *= sigm(acc[t][r] + bb);
  }
  ZERO(); GEMM(x, W_all + 4 * HH);
#pragma unroll
  for (int t = 0; t < NT; ++t) {
    float bb = b_all[4 * HD + c.colBase + t * 16 + c.colLo];
#pragma unroll
    for (int r = 0; r < 8; ++r) keepB[t][r] = acc[t][r] + bb;
  }
  ZERO(); GEMM(h_prev, U_all + 4 * HH);
#pragma unroll
  for (int t = 0; t < NT; ++t)
#pragma unroll
    for (int r = 0; r < 8; ++r) keepB[t][r] *= acc[t][r];
  ZERO(); GEMM(x, W_all + 5 * HH); GEMM(h_prev, U_all + 5 * HH);
#pragma unroll
  for (int t = 0; t < NT; ++t) {
    float bb = b_all[5 * HD + c.colBase + t * 16 + c.colLo];
#pragma unroll
    for (int r = 0; r < 8; ++r) keepA[t][r] += sigm(acc[t][r] + bb) * keepB[t][r];
  }
  ZERO(); GEMM(x, W_all + HH); GEMM(h_prev, U_all + HH);
#pragma unroll
  for (int t = 0; t < NT; ++t) {
    float bb = b_all[HD + c.colBase + t * 16 + c.colLo];
#pragma unroll
    for (int r = 0; r < 8; ++r) {
      size_t idx = (size_t)(c.mBase + c.rowOff + r) * HD + (c.colBase + t * 16 + c.colLo);
      keepA[t][r] += sigm(acc[t][r] + bb) * c_prev[idx];
      out[(size_t)B_DIM * HD + idx] = keepA[t][r];
    }
  }
  ZERO(); GEMM(x, W_all + 2 * HH); GEMM(h_prev, U_all + 2 * HH);
#pragma unroll
  for (int t = 0; t < NT; ++t) {
    float bb = b_all[2 * HD + c.colBase + t * 16 + c.colLo];
#pragma unroll
    for (int r = 0; r < 8; ++r) {
      size_t idx = (size_t)(c.mBase + c.rowOff + r) * HD + (c.colBase + t * 16 + c.colLo);
      out[idx] = sigm(acc[t][r] + bb) * tanhf(keepA[t][r]);
    }
  }
#undef GEMM
#undef ZERO
}

extern "C" void kernel_launch(void* const* d_in, const int* in_sizes, int n_in,
                              void* d_out, int out_size, void* d_ws, size_t ws_size,
                              hipStream_t stream) {
  const float* x      = (const float*)d_in[0];
  const float* h_prev = (const float*)d_in[1];
  const float* c_prev = (const float*)d_in[2];
  const float* W_all  = (const float*)d_in[3];
  const float* b_all  = (const float*)d_in[4];
  const float* U_all  = (const float*)d_in[5];
  float* out = (float*)d_out;

  dim3 grid(B_DIM / BM, HD / BN);   // 32 x 8 = 256 blocks

  const size_t nXH = (size_t)B_DIM * HD;                      // per x / h
  const size_t nW  = (size_t)6 * HH;                          // per W / U
  const size_t need = (2 * nXH + 2 * nW) * sizeof(__bf16);    // 41.9 MB

  if (ws_size >= need) {
    __bf16* A2 = (__bf16*)d_ws;           // [4096, 2048] = [x | h]
    __bf16* Wc = A2 + 2 * nXH;            // 6 panels [1024, 2048] = [W_k | U_k]
    const int thr = 256;
    cvt_pack<<<(int)((nXH / 8 + thr - 1) / thr), thr, 0, stream>>>(x,      A2, (int)(nXH / 8), 0);
    cvt_pack<<<(int)((nXH / 8 + thr - 1) / thr), thr, 0, stream>>>(h_prev, A2, (int)(nXH / 8), HD);
    cvt_pack<<<(int)((nW  / 8 + thr - 1) / thr), thr, 0, stream>>>(W_all,  Wc, (int)(nW  / 8), 0);
    cvt_pack<<<(int)((nW  / 8 + thr - 1) / thr), thr, 0, stream>>>(U_all,  Wc, (int)(nW  / 8), HD);
    xlstm_packed<<<grid, THREADS, 0, stream>>>(A2, c_prev, Wc, b_all, out);
  } else {
    xlstm_f32<<<grid, THREADS, 0, stream>>>(x, h_prev, c_prev, W_all, b_all, U_all, out);
  }
}
